// NNUE_26955214750206
// MI455X (gfx1250) — matrix operand order repaired and checked by reading, not compiled, not run on hardware
//
#include <hip/hip_runtime.h>
#include <hip/hip_bf16.h>

typedef __attribute__((ext_vector_type(16))) __bf16 v16bf;
typedef __attribute__((ext_vector_type(8)))  float  v8f;

// -------------------------------------------------------------------------
// Phase 1: segmented embedding accumulate (bandwidth-critical, ~512MB read).
// One block per (batch, side). batch array is sorted -> binary search range.
// 256 threads = one f32 column each; rows are 1KB fully-coalesced loads.
// Output written as bf16 directly into the flip-resolved x[B,512] buffer.
// -------------------------------------------------------------------------
__global__ void nnue_accum_kernel(const int* __restrict__ idxArr,
                                  const int* __restrict__ batchArr,
                                  const int* __restrict__ stm,
                                  const float* __restrict__ emb,
                                  __bf16* __restrict__ x,
                                  int nAct, int side)
{
    const int b   = blockIdx.x;
    const int tid = threadIdx.x;            // column 0..255

    // lower_bound(b)
    int lo = 0, hi = nAct;
    while (lo < hi) { int mid = (lo + hi) >> 1; if (batchArr[mid] <  b) lo = mid + 1; else hi = mid; }
    const int start = lo;
    // upper_bound(b)
    hi = nAct;
    while (lo < hi) { int mid = (lo + hi) >> 1; if (batchArr[mid] <= b) lo = mid + 1; else hi = mid; }
    const int end = lo;

    float acc = 0.0f;
    if (start < end) {
        int f = idxArr[start];
        for (int r = start; r < end; ++r) {
            const int fn = (r + 1 < end) ? idxArr[r + 1] : f;
            // hide index->row latency: prefetch next embedding row (global_prefetch_b8)
            __builtin_prefetch(emb + (size_t)fn * 256 + tid, 0, 0);
            acc += emb[(size_t)f * 256 + tid];
            f = fn;
        }
    }

    const int flip = (stm[b] == 0);
    const int base = (side == 0) ? (flip ? 256 : 0) : (flip ? 0 : 256);
    x[(size_t)b * 512 + base + tid] = (__bf16)acc;
}

// -------------------------------------------------------------------------
// Repack W (K x N, f32 row-major) into WMMA B-operand tile layout (bf16):
// tile index t = nTile*kTiles + kt; within a tile: 32 lanes x 16 elements,
// element j of lane L is B[k = kt*32 + (L>=16?16:0) + j][n = nTile*16 + (L&15)]
// so each lane's 16 B values are one contiguous, 32B-aligned load in the GEMM.
// -------------------------------------------------------------------------
__global__ void nnue_packw_kernel(const float* __restrict__ W,
                                  __bf16* __restrict__ Bp,
                                  int K, int N)
{
    const int idx = blockIdx.x * blockDim.x + threadIdx.x;
    if (idx >= K * N) return;
    const int j      = idx & 15;
    const int lane   = (idx >> 4) & 31;
    const int tile   = idx >> 9;
    const int kTiles = K >> 5;
    const int nTile  = tile / kTiles;
    const int kt     = tile % kTiles;
    const int krow   = kt * 32 + ((lane >= 16) ? 16 : 0) + j;
    const int ncol   = nTile * 16 + (lane & 15);
    Bp[idx] = (__bf16)W[(size_t)krow * N + ncol];
}

// -------------------------------------------------------------------------
// WMMA GEMM: out = relu(A[MxK] * W[KxN] + bias), A bf16 row-major,
// W pre-packed bf16 tiles, f32 accumulate. One wave per 16x16 output tile.
// Grid sized exactly (M,N multiples of 16) so EXEC is all-1s at every WMMA.
// -------------------------------------------------------------------------
__global__ void nnue_wmma_gemm_kernel(const __bf16* __restrict__ A,
                                      const __bf16* __restrict__ Bp,
                                      const float* __restrict__ bias,
                                      void* __restrict__ out,
                                      int M, int K, int N, int outBf16)
{
    const int lane   = threadIdx.x & 31;
    const int gwave  = blockIdx.x * (blockDim.x >> 5) + (threadIdx.x >> 5);
    const int nTiles = N >> 4;
    const int mTile  = gwave / nTiles;
    const int nTile  = gwave % nTiles;
    if (mTile * 16 >= M) return;              // wave-uniform: EXEC stays full

    const int kTiles = K >> 5;
    const int m      = mTile * 16 + (lane & 15);
    const int khalf  = (lane >> 4) << 4;      // 0 or 16 (A layout half-select)
    const __bf16* __restrict__ arow = A  + (size_t)m * K;
    const __bf16* __restrict__ brow = Bp + (size_t)nTile * kTiles * 512 + lane * 16;

    v8f acc = {};
    for (int kt = 0; kt < kTiles; ++kt) {
        v16bf a, bm;
        const __bf16* ak = arow + kt * 32 + khalf;   // 16B-aligned pairs
        const __bf16* bk = brow + kt * 512;          // 32B-aligned, contiguous
#pragma unroll
        for (int j = 0; j < 8; ++j) { a[j] = ak[j]; a[8 + j] = ak[8 + j]; }
#pragma unroll
        for (int j = 0; j < 16; ++j) bm[j] = bk[j];
        acc = __builtin_amdgcn_wmma_f32_16x16x32_bf16(
            /*neg_a=*/false, a, /*neg_b=*/false, bm,
            /*c_mod=*/(short)0, acc, /*reuse_a=*/false, /*reuse_b=*/false);
    }

    // D layout: VGPR i -> M = i (lanes 0-15) / M = 8+i (lanes 16-31)
    const int nCol    = nTile * 16 + (lane & 15);
    const int rowBase = mTile * 16 + ((lane >> 4) << 3);
    const float bv = bias[nCol];
#pragma unroll
    for (int i = 0; i < 8; ++i) {
        float v = acc[i] + bv;
        v = v > 0.0f ? v : 0.0f;
        const size_t o = (size_t)(rowBase + i) * N + nCol;
        if (outBf16) ((__bf16*)out)[o] = (__bf16)v;
        else         ((float*)out)[o]  = v;
    }
}

// -------------------------------------------------------------------------
// Tail: per-row fused relu(h2@W3+b3)@Wo+bo  (64->32->1, trivial FLOPs).
// -------------------------------------------------------------------------
__global__ void nnue_tail_kernel(const float* __restrict__ h2,
                                 const float* __restrict__ W3,
                                 const float* __restrict__ b3,
                                 const float* __restrict__ Wo,
                                 const float* __restrict__ bo,
                                 float* __restrict__ out, int B)
{
    const int row = blockIdx.x * blockDim.x + threadIdx.x;
    if (row >= B) return;
    float h[64];
    const float* hr = h2 + (size_t)row * 64;
#pragma unroll
    for (int i = 0; i < 64; ++i) h[i] = hr[i];
    float accO = 0.0f;
#pragma unroll 4
    for (int j = 0; j < 32; ++j) {
        float s = b3[j];
#pragma unroll
        for (int i = 0; i < 64; ++i) s += h[i] * W3[i * 32 + j];
        s = s > 0.0f ? s : 0.0f;
        accO += s * Wo[j];
    }
    out[row] = accO + bo[0];
}

static inline size_t alignUp(size_t v, size_t a) { return (v + a - 1) & ~(a - 1); }

extern "C" void kernel_launch(void* const* d_in, const int* in_sizes, int n_in,
                              void* d_out, int out_size, void* d_ws, size_t ws_size,
                              hipStream_t stream)
{
    const int*   white_idx   = (const int*)  d_in[0];
    const int*   black_idx   = (const int*)  d_in[1];
    const int*   white_batch = (const int*)  d_in[2];
    const int*   black_batch = (const int*)  d_in[3];
    const int*   stm         = (const int*)  d_in[4];
    const float* white_emb   = (const float*)d_in[5];
    const float* black_emb   = (const float*)d_in[6];
    const float* W1          = (const float*)d_in[7];
    const float* b1          = (const float*)d_in[8];
    const float* W2          = (const float*)d_in[9];
    const float* b2          = (const float*)d_in[10];
    const float* W3          = (const float*)d_in[11];
    const float* b3          = (const float*)d_in[12];
    const float* Wo          = (const float*)d_in[13];
    const float* bo          = (const float*)d_in[14];

    const int B     = in_sizes[4];     // 8192
    const int nActW = in_sizes[0];     // 262144
    const int nActB = in_sizes[1];

    // workspace carve-out
    char*  ws  = (char*)d_ws;
    size_t off = 0;
    __bf16* x   = (__bf16*)(ws + off); off = alignUp(off + (size_t)B * 512 * sizeof(__bf16), 256);
    __bf16* h1  = (__bf16*)(ws + off); off = alignUp(off + (size_t)B * 128 * sizeof(__bf16), 256);
    float*  h2  = (float*) (ws + off); off = alignUp(off + (size_t)B * 64  * sizeof(float),  256);
    __bf16* pW1 = (__bf16*)(ws + off); off = alignUp(off + (size_t)512 * 128 * sizeof(__bf16), 256);
    __bf16* pW2 = (__bf16*)(ws + off); off = alignUp(off + (size_t)128 * 64  * sizeof(__bf16), 256);
    (void)ws_size; (void)n_in; (void)out_size;

    // pack W1, W2 into WMMA tile layout (tiny)
    nnue_packw_kernel<<<(512 * 128 + 255) / 256, 256, 0, stream>>>(W1, pW1, 512, 128);
    nnue_packw_kernel<<<(128 * 64  + 255) / 256, 256, 0, stream>>>(W2, pW2, 128, 64);

    // gather + segment-sum, flip folded into store (dominant ~512MB phase)
    nnue_accum_kernel<<<B, 256, 0, stream>>>(white_idx, white_batch, stm, white_emb, x, nActW, 0);
    nnue_accum_kernel<<<B, 256, 0, stream>>>(black_idx, black_batch, stm, black_emb, x, nActB, 1);

    // layer 1: [B,512]x[512,128] -> bf16 h1 ; (B/16)*(128/16) waves, 8 waves/block
    {
        const int waves = (B / 16) * (128 / 16);
        nnue_wmma_gemm_kernel<<<waves / 8, 256, 0, stream>>>(x, pW1, b1, h1, B, 512, 128, 1);
    }
    // layer 2: [B,128]x[128,64] -> f32 h2
    {
        const int waves = (B / 16) * (64 / 16);
        nnue_wmma_gemm_kernel<<<waves / 8, 256, 0, stream>>>(h1, pW2, b2, h2, B, 128, 64, 0);
    }
    // layers 3+4 fused
    nnue_tail_kernel<<<(B + 127) / 128, 128, 0, stream>>>(h2, W3, b3, Wo, bo, (float*)d_out, B);
}
